// LinearRegressor_4913442587015
// MI455X (gfx1250) — matrix operand order repaired and checked by reading, not compiled, not run on hardware
//
#include <hip/hip_runtime.h>

#define NF     8192
#define BATCHN 64
#define NNZ    (9 * NF)
#define KSPLIT 8

typedef __attribute__((ext_vector_type(16))) _Float16 v16h;
typedef __attribute__((ext_vector_type(8)))  float    v8f;
typedef __attribute__((ext_vector_type(4)))  float    fv4;

// ---------------------------------------------------------------------------
// Zero kernels
// ---------------------------------------------------------------------------
__global__ void fea_zero1_kernel(float* __restrict__ h) {
    int t = blockIdx.x * blockDim.x + threadIdx.x;     // 0..131071 (fv4)
    fv4 z = {0.f, 0.f, 0.f, 0.f};
    ((fv4*)h)[t] = z;
}

__global__ void fea_zero2_kernel(float* __restrict__ h, float* __restrict__ out) {
    int t = blockIdx.x * blockDim.x + threadIdx.x;     // 0..131071 (fv4)
    fv4 z = {0.f, 0.f, 0.f, 0.f};
    ((fv4*)h)[t]   = z;
    ((fv4*)out)[t] = z;
}

// ---------------------------------------------------------------------------
// K1: h[b, rows[e]] += vals[e] * x[b, cols[e]]
// rows are sorted -> merge runs of equal rows locally before atomicAdd.
// thread = 8-entry chunk of nnz, blockIdx.y = batch row.
// ---------------------------------------------------------------------------
__global__ void fea_spmv_kernel(const float* __restrict__ x,
                                const int*   __restrict__ rows,
                                const int*   __restrict__ cols,
                                const float* __restrict__ vals,
                                float* __restrict__ h) {
    int chunk = blockIdx.x * blockDim.x + threadIdx.x;   // 0..9215
    int b     = blockIdx.y;                              // 0..63
    int e0    = chunk * 8;
    const float* xb = x + (size_t)b * NF;
    float*       hb = h + (size_t)b * NF;

    int   cur = rows[e0];
    float acc = 0.f;
#pragma unroll
    for (int i = 0; i < 8; ++i) {
        int   e = e0 + i;
        int   r = rows[e];
        float v = vals[e] * xb[cols[e]];
        if (r != cur) {
            atomicAdd(hb + cur, acc);
            cur = r;
            acc = v;
        } else {
            acc += v;
        }
    }
    atomicAdd(hb + cur, acc);
}

// ---------------------------------------------------------------------------
// K2: AhT[k][b] = f16( bc[k] + h[b][k] * flag[k] )   (transposed f16 store,
// K-major so it matches the WMMA B-matrix layout: lane=N, 16 K values/lane)
// ---------------------------------------------------------------------------
__global__ void fea_addcmul_t_kernel(const float* __restrict__ h,
                                     const float* __restrict__ bc,
                                     const float* __restrict__ flag,
                                     _Float16* __restrict__ AhT) {
    int t = blockIdx.x * blockDim.x + threadIdx.x;       // 0..524287
    int b = t & (BATCHN - 1);
    int k = t >> 6;
    float v = bc[k] + h[(size_t)b * NF + k] * flag[k];
    AhT[(size_t)k * BATCHN + b] = (_Float16)v;
}

// ---------------------------------------------------------------------------
// K3: GEMM  out[b][i] += sum_k Ah[b][k] * W[i][k]
//   WMMA roles: A = 16-feature tile of W (f32 -> f16 in regs, nontemporal),
//               B = AhT tile (f16, K-major, L2-hot), C/D = f32 accum.
//   Per wave: 1 feature tile (M), 4 batch tiles (N), K-chunk of 1024.
// ---------------------------------------------------------------------------
#define CVT4(dst, i0, v)                          \
    dst[(i0) + 0] = (_Float16)(v).x;              \
    dst[(i0) + 1] = (_Float16)(v).y;              \
    dst[(i0) + 2] = (_Float16)(v).z;              \
    dst[(i0) + 3] = (_Float16)(v).w;

// Shared GEMM body: computes the 4 accumulators for this wave.
__device__ __forceinline__ void fea_gemm_body(const float* __restrict__ W,
                                              const _Float16* __restrict__ AhT,
                                              int f0, int lrow, int hsel, int k0,
                                              v8f& acc0, v8f& acc1, v8f& acc2, v8f& acc3) {
    const float* Wr = W + (size_t)(f0 + lrow) * NF;
    const int lane = lrow + hsel * 16;

    for (int k = k0; k < k0 + NF / KSPLIT; k += 32) {
        // ---- A fragment: 16x32 f16 tile of W.
        // lane(hsel=0): K k+0..7 and k+16..23 ; lane(hsel=1): K k+8..15 and k+24..31
        const int ka = k + hsel * 8;
        fv4 w0 = __builtin_nontemporal_load((const fv4*)(Wr + ka));
        fv4 w1 = __builtin_nontemporal_load((const fv4*)(Wr + ka) + 1);
        fv4 w2 = __builtin_nontemporal_load((const fv4*)(Wr + ka + 16));
        fv4 w3 = __builtin_nontemporal_load((const fv4*)(Wr + ka + 16) + 1);
        v16h a;
        CVT4(a,  0, w0); CVT4(a,  4, w1);
        CVT4(a,  8, w2); CVT4(a, 12, w3);

        // ---- B fragments: lane holds batch col lrow of each tile, K in same order as A
        v16h b0v, b1v, b2v, b3v;
        for (int s = 0; s < 16; ++s) {
            const _Float16* Br = AhT + (size_t)(k + hsel * 8 + (s & 7) + (s >> 3) * 16) * BATCHN + lrow;
            b0v[s] = Br[0];
            b1v[s] = Br[16];
            b2v[s] = Br[32];
            b3v[s] = Br[48];
        }

        acc0 = __builtin_amdgcn_wmma_f32_16x16x32_f16(false, a, false, b0v,
                                                      (short)0, acc0, false, false);
        acc1 = __builtin_amdgcn_wmma_f32_16x16x32_f16(false, a, false, b1v,
                                                      (short)0, acc1, false, false);
        acc2 = __builtin_amdgcn_wmma_f32_16x16x32_f16(false, a, false, b2v,
                                                      (short)0, acc2, false, false);
        acc3 = __builtin_amdgcn_wmma_f32_16x16x32_f16(false, a, false, b3v,
                                                      (short)0, acc3, false, false);
    }
}

// Variant A (preferred): plain vector stores of per-K-chunk partials.
// D layout: lane col n = lrow (batch in tile), VGPR r -> m = hsel*8 + r, so each
// accumulator is 8 CONSECUTIVE floats along the feature dim -> two b128 stores.
__global__ void __launch_bounds__(128)
fea_gemm_part_kernel(const float* __restrict__ W,
                     const _Float16* __restrict__ AhT,
                     float* __restrict__ partials) {
    const int lane = threadIdx.x & 31;
    const int wave = threadIdx.x >> 5;
    const int lrow = lane & 15;
    const int hsel = lane >> 4;
    const int f0   = (blockIdx.x * 4 + wave) * 16;
    const int k0   = blockIdx.y * (NF / KSPLIT);

    v8f acc0 = {}, acc1 = {}, acc2 = {}, acc3 = {};
    fea_gemm_body(W, AhT, f0, lrow, hsel, k0, acc0, acc1, acc2, acc3);

    float* P = partials + (size_t)blockIdx.y * BATCHN * NF;
    const int i0 = f0 + hsel * 8;
    fv4* p0 = (fv4*)(P + (size_t)( 0 + lrow) * NF + i0);
    fv4* p1 = (fv4*)(P + (size_t)(16 + lrow) * NF + i0);
    fv4* p2 = (fv4*)(P + (size_t)(32 + lrow) * NF + i0);
    fv4* p3 = (fv4*)(P + (size_t)(48 + lrow) * NF + i0);
    p0[0] = *(fv4*)&acc0;  p0[1] = *((fv4*)&acc0 + 1);
    p1[0] = *(fv4*)&acc1;  p1[1] = *((fv4*)&acc1 + 1);
    p2[0] = *(fv4*)&acc2;  p2[1] = *((fv4*)&acc2 + 1);
    p3[0] = *(fv4*)&acc3;  p3[1] = *((fv4*)&acc3 + 1);
}

// Variant B (fallback if ws too small): atomic accumulation into out.
__global__ void __launch_bounds__(128)
fea_gemm_atomic_kernel(const float* __restrict__ W,
                       const _Float16* __restrict__ AhT,
                       float* __restrict__ out) {
    const int lane = threadIdx.x & 31;
    const int wave = threadIdx.x >> 5;
    const int lrow = lane & 15;
    const int hsel = lane >> 4;
    const int f0   = (blockIdx.x * 4 + wave) * 16;
    const int k0   = blockIdx.y * (NF / KSPLIT);

    v8f acc0 = {}, acc1 = {}, acc2 = {}, acc3 = {};
    fea_gemm_body(W, AhT, f0, lrow, hsel, k0, acc0, acc1, acc2, acc3);

    const int i0 = f0 + hsel * 8;
#pragma unroll
    for (int r = 0; r < 8; ++r) {
        atomicAdd(out + (size_t)( 0 + lrow) * NF + (i0 + r), acc0[r]);
        atomicAdd(out + (size_t)(16 + lrow) * NF + (i0 + r), acc1[r]);
        atomicAdd(out + (size_t)(32 + lrow) * NF + (i0 + r), acc2[r]);
        atomicAdd(out + (size_t)(48 + lrow) * NF + (i0 + r), acc3[r]);
    }
}

// ---------------------------------------------------------------------------
// K4a: reduce KSPLIT partial slices (L2-hot) + fused addcmul epilogue
// ---------------------------------------------------------------------------
__global__ void fea_reduce_epilogue_kernel(const float* __restrict__ partials,
                                           const float* __restrict__ bc,
                                           const float* __restrict__ flag,
                                           float* __restrict__ out) {
    int t = blockIdx.x * blockDim.x + threadIdx.x;       // 0..524287
    int i = t & (NF - 1);
    float s = 0.f;
#pragma unroll
    for (int c = 0; c < KSPLIT; ++c)
        s += partials[(size_t)c * BATCHN * NF + t];
    out[t] = bc[i] + s * flag[i];
}

// K4b: epilogue only (atomic fallback path)
__global__ void fea_epilogue_kernel(float* __restrict__ out,
                                    const float* __restrict__ bc,
                                    const float* __restrict__ flag) {
    int t = blockIdx.x * blockDim.x + threadIdx.x;       // 0..524287
    int i = t & (NF - 1);
    out[t] = bc[i] + out[t] * flag[i];
}

// ---------------------------------------------------------------------------
extern "C" void kernel_launch(void* const* d_in, const int* in_sizes, int n_in,
                              void* d_out, int out_size, void* d_ws, size_t ws_size,
                              hipStream_t stream) {
    (void)in_sizes; (void)n_in; (void)out_size;

    const float* x    = (const float*)d_in[0];   // [64, 8192]
    const int*   rows = (const int*)  d_in[1];   // [73728] sorted
    const int*   cols = (const int*)  d_in[2];   // [73728]
    const float* vals = (const float*)d_in[3];   // [73728]
    const float* bc   = (const float*)d_in[4];   // [8192]
    const float* flag = (const float*)d_in[5];   // [8192]
    const float* W    = (const float*)d_in[6];   // [8192, 8192]
    float*       out  = (float*)d_out;           // [64, 8192]

    const size_t hBytes    = (size_t)BATCHN * NF * sizeof(float);     // 2 MB
    const size_t ahtBytes  = (size_t)BATCHN * NF * sizeof(_Float16);  // 1 MB
    const size_t partBytes = (size_t)KSPLIT * BATCHN * NF * sizeof(float); // 16 MB

    float*    h   = (float*)d_ws;
    _Float16* AhT = (_Float16*)((char*)d_ws + hBytes);

    const bool usePartials = (ws_size >= hBytes + ahtBytes + partBytes);

    if (usePartials) {
        float* partials = (float*)((char*)d_ws + hBytes + ahtBytes);

        fea_zero1_kernel<<<512, 256, 0, stream>>>(h);
        fea_spmv_kernel<<<dim3(72, 64), 128, 0, stream>>>(x, rows, cols, vals, h);
        fea_addcmul_t_kernel<<<2048, 256, 0, stream>>>(h, bc, flag, AhT);
        fea_gemm_part_kernel<<<dim3(128, KSPLIT), 128, 0, stream>>>(W, AhT, partials);
        fea_reduce_epilogue_kernel<<<2048, 256, 0, stream>>>(partials, bc, flag, out);
    } else {
        fea_zero2_kernel<<<512, 256, 0, stream>>>(h, out);
        fea_spmv_kernel<<<dim3(72, 64), 128, 0, stream>>>(x, rows, cols, vals, h);
        fea_addcmul_t_kernel<<<2048, 256, 0, stream>>>(h, bc, flag, AhT);
        fea_gemm_atomic_kernel<<<dim3(128, KSPLIT), 128, 0, stream>>>(W, AhT, out);
        fea_epilogue_kernel<<<2048, 256, 0, stream>>>(out, bc, flag);
    }
}
